// RNNModel_25108378812624
// MI455X (gfx1250) — matrix operand order repaired and checked by reading, not compiled, not run on hardware
//
#include <hip/hip_runtime.h>

// Fused fc0 + LSTM(T=512) + head for MI455X (gfx1250, wave32, WMMA).
// One workgroup (128 thr = 4 waves) owns a 16-row batch tile for all timesteps.
// Each wave owns ALL FOUR gates for one 16-wide hidden tile, so the cell update
// runs directly on the WMMA accumulators in registers (no gate LDS round-trip).
// Double-buffered h/h0 in LDS -> exactly ONE barrier per timestep.

#define IN_F   5
#define D_FC   32
#define D_H    64
#define T_LEN  512
#define BTILE  16

typedef __attribute__((ext_vector_type(16))) __bf16 v16bf;
typedef __attribute__((ext_vector_type(8)))  __bf16 v8bf;
typedef __attribute__((ext_vector_type(4)))  __bf16 v4bf;
typedef __attribute__((ext_vector_type(8)))  float  v8f;

#define WMMA_BF16(A, B, C) \
  __builtin_amdgcn_wmma_f32_16x16x32_bf16(false, (A), false, (B), (short)0, (C), false, false)

#define LOG2E_F 1.4426950408889634f

// Branch-free sigmoid: 1/(1 + 2^(-x*log2e)) -> v_exp_f32 + v_rcp_f32.
static __device__ inline float fast_sigmoid(float x) {
  return __builtin_amdgcn_rcpf(1.0f + __builtin_amdgcn_exp2f(-LOG2E_F * x));
}

// Branch-free tanh: native v_tanh_f32 on gfx1250, exp2 identity otherwise.
static __device__ inline float fast_tanh(float x) {
#if __has_builtin(__builtin_amdgcn_tanhf)
  return __builtin_amdgcn_tanhf(x);
#else
  return 1.0f - 2.0f * __builtin_amdgcn_rcpf(1.0f + __builtin_amdgcn_exp2f(2.0f * LOG2E_F * x));
#endif
}

// B-operand fragment (32x16, K x N): lane l holds column n = l&15; its 16
// elements are K = kbase + (l>>4)*8 + e (e<8) and kbase + 16 + (l>>4)*8 + (e-8) (e>=8).
static __device__ inline v16bf load_b_frag_f32(const float* __restrict__ W, int ldk,
                                               int ntile, int kbase, int lane) {
  const int n = lane & 15;
  const int khalf = lane >> 4;
  const float* p = W + (size_t)(ntile * 16 + n) * ldk + kbase + khalf * 8;
  v16bf r;
#pragma unroll
  for (int e = 0; e < 16; ++e) r[e] = (__bf16)p[e < 8 ? e : e + 8];
  return r;
}

// A-operand fragment (16x32, M x K) from row-major bf16 LDS tile.
// Lanes 0-15 (row m): K = 0..7 and 16..23; lanes 16-31: K = 8..15 and 24..31
// -> two contiguous 16-byte LDS loads per lane.
static __device__ inline v16bf load_a_frag_lds(const __bf16* base, int ldk,
                                               int kbase, int lane) {
  const int m = lane & 15;
  const int sel = lane >> 4;
  const __bf16* p = base + m * ldk + kbase + sel * 8;
  v8bf lo = *(const v8bf*)(p);
  v8bf hi = *(const v8bf*)(p + 16);
  return __builtin_shufflevector(lo, hi, 0, 1, 2, 3, 4, 5, 6, 7,
                                 8, 9, 10, 11, 12, 13, 14, 15);
}

__global__ __launch_bounds__(128)
void lstm_fused_kernel(const float* __restrict__ x,
                       const float* __restrict__ W0,
                       const float* __restrict__ b0,
                       const float* __restrict__ W_ih,
                       const float* __restrict__ W_hh,
                       const float* __restrict__ b_ih,
                       const float* __restrict__ b_hh,
                       const float* __restrict__ Wo,
                       const float* __restrict__ bo,
                       float* __restrict__ out)
{
  __shared__ __bf16 lds_h[2][BTILE * D_H];    // 2x2 KB: hidden state, dbl-buffered
  __shared__ __bf16 lds_h0[2][BTILE * D_FC];  // 2x1 KB: fc0 output, dbl-buffered
  __shared__ float  lds_hf[BTILE * D_H];      //   4 KB: fp32 h at final step

  const int tid  = threadIdx.x;
  const int lane = tid & 31;
  const int wave = tid >> 5;                  // 0..3 = hidden tile index
  const int row0 = blockIdx.x * BTILE;

  // zero the buffer read at t=0 (h(-1) lives in buffer 1)
  for (int i = tid; i < BTILE * D_H; i += 128) lds_h[1][i] = (__bf16)0.0f;

  // ---- per-wave weight fragments (12), pinned in VGPRs for all 512 steps ----
  // gate g (i,f,g,o) x this wave's hidden tile -> ntile = g*4 + wave
  v16bf wih[4], whh0[4], whh1[4];
#pragma unroll
  for (int g = 0; g < 4; ++g) {
    const int nt = g * 4 + wave;
    wih[g]  = load_b_frag_f32(W_ih, D_FC, nt, 0,  lane);
    whh0[g] = load_b_frag_f32(W_hh, D_H,  nt, 0,  lane);
    whh1[g] = load_b_frag_f32(W_hh, D_H,  nt, 32, lane);
  }

  // per-lane gate bias: accumulator column is shared by all 8 rows of a lane
  const int coln = wave * 16 + (lane & 15);   // hidden column 0..63
  const float bi  = b_ih[0 * D_H + coln] + b_hh[0 * D_H + coln];
  const float bf_ = b_ih[1 * D_H + coln] + b_hh[1 * D_H + coln];
  const float bg  = b_ih[2 * D_H + coln] + b_hh[2 * D_H + coln];
  const float bo_ = b_ih[3 * D_H + coln] + b_hh[3 * D_H + coln];

  // ---- fc0 assignment: 4 output channels per thread ----
  const int ra = tid >> 3;                    // batch row 0..15
  const int c4 = (tid & 7) * 4;               // fc0 channel base 0..28
  float w0r[4][IN_F], b0r[4];
#pragma unroll
  for (int q = 0; q < 4; ++q) {
#pragma unroll
    for (int j = 0; j < IN_F; ++j) w0r[q][j] = W0[(c4 + q) * IN_F + j];
    b0r[q] = b0[c4 + q];
  }

  // cell state: 8 rows per lane (rows rb..rb+7, column coln), in registers
  const int rb = (lane < 16) ? 0 : 8;
  float cst[8];
#pragma unroll
  for (int v = 0; v < 8; ++v) cst[v] = 0.0f;

  // x stream for this thread's fc0 row; software-pipelined
  const float* xrow = x + (size_t)(row0 + ra) * T_LEN * IN_F;
  float xv[IN_F];
#pragma unroll
  for (int j = 0; j < IN_F; ++j) xv[j] = xrow[j];

  for (int t = 0; t < T_LEN; ++t) {
    const int wb = t & 1;                     // write buffer
    const int rd = wb ^ 1;                    // read buffer (h(t-1))

    // ---- Phase A: fc0 -> lds_h0[wb] (bf16, row-major) ----
    v4bf hp;
#pragma unroll
    for (int q = 0; q < 4; ++q) {
      float s = b0r[q];
#pragma unroll
      for (int j = 0; j < IN_F; ++j) s += xv[j] * w0r[q][j];
      hp[q] = (__bf16)s;
    }
    *(v4bf*)&lds_h0[wb][ra * D_FC + c4] = hp;

    __syncthreads();                          // the ONLY barrier per step

    // ---- Phase B: 4 independent 3-WMMA chains (i, f, g, o) ----
    v16bf a0  = load_a_frag_lds(lds_h0[wb], D_FC, 0,  lane);
    v16bf ah0 = load_a_frag_lds(lds_h[rd],  D_H,  0,  lane);
    v16bf ah1 = load_a_frag_lds(lds_h[rd],  D_H,  32, lane);

    v8f acc[4];
#pragma unroll
    for (int g = 0; g < 4; ++g) {
      v8f a = {};
      a = WMMA_BF16(a0,  wih[g],  a);
      a = WMMA_BF16(ah0, whh0[g], a);
      a = WMMA_BF16(ah1, whh1[g], a);
      acc[g] = a;
    }

    // prefetch next timestep's x while WMMAs drain
    if (t + 1 < T_LEN) {
#pragma unroll
      for (int j = 0; j < IN_F; ++j) xv[j] = xrow[(size_t)(t + 1) * IN_F + j];
    }

    // ---- Phase C: cell update directly on accumulators (registers only) ----
#pragma unroll
    for (int v = 0; v < 8; ++v) {
      float gi = fast_sigmoid(acc[0][v] + bi);
      float gf = fast_sigmoid(acc[1][v] + bf_);
      float gg = fast_tanh   (acc[2][v] + bg);
      float go = fast_sigmoid(acc[3][v] + bo_);
      float c  = gf * cst[v] + gi * gg;
      cst[v] = c;
      float h = go * fast_tanh(c);
      lds_h[wb][(rb + v) * D_H + coln] = (__bf16)h;
      if (t == T_LEN - 1) lds_hf[(rb + v) * D_H + coln] = h;
    }
  }
  __syncthreads();

  // ---- head: out[row] = h_last . Wo + bo (fp32 h) ----
  if (tid < BTILE) {
    float s = bo[0];
#pragma unroll
    for (int k = 0; k < D_H; ++k) s += lds_hf[tid * D_H + k] * Wo[k];
    out[row0 + tid] = s;
  }
}

extern "C" void kernel_launch(void* const* d_in, const int* in_sizes, int n_in,
                              void* d_out, int out_size, void* d_ws, size_t ws_size,
                              hipStream_t stream) {
  (void)in_sizes; (void)n_in; (void)d_ws; (void)ws_size;
  const float* x    = (const float*)d_in[0];
  const float* W0   = (const float*)d_in[1];
  const float* b0   = (const float*)d_in[2];
  const float* W_ih = (const float*)d_in[3];
  const float* W_hh = (const float*)d_in[4];
  const float* b_ih = (const float*)d_in[5];
  const float* b_hh = (const float*)d_in[6];
  const float* Wo   = (const float*)d_in[7];
  const float* bo   = (const float*)d_in[8];
  float* out = (float*)d_out;

  const int B = out_size;                 // 4096 rows, [B,1] output
  dim3 grid(B / BTILE), block(128);
  hipLaunchKernelGGL(lstm_fused_kernel, grid, block, 0, stream,
                     x, W0, b0, W_ih, W_hh, b_ih, b_hh, Wo, bo, out);
}
